// Discriminator_60902636257998
// MI455X (gfx1250) — matrix operand order repaired and checked by reading, not compiled, not run on hardware
//
#include <hip/hip_runtime.h>
#include <hip/hip_bf16.h>

#define B_ 64
#define S_ 512
#define E_ 512
#define H_ 1024

typedef __attribute__((ext_vector_type(16))) __bf16 v16bf;
typedef __attribute__((ext_vector_type(8)))  float  v8f;

struct alignas(32) Pack32 { uint4 a, b; };

static __device__ __forceinline__ unsigned short f2bf(float f) {
    unsigned u = __builtin_bit_cast(unsigned, f);
    unsigned r = u + 0x7FFFu + ((u >> 16) & 1u);   // round-to-nearest-even
    return (unsigned short)(r >> 16);
}

static __device__ __forceinline__ float sigmoidf_(float x) {
    return 1.0f / (1.0f + __expf(-x));
}

// A fragment: 16x32 bf16, row-major source, per-lane row = r0 + (lane&15)
// elements 0..15 -> K = k0 + 16*hi + e
static __device__ __forceinline__ v16bf load_a(const unsigned short* __restrict__ row,
                                               int k0, int hi) {
    Pack32 p;
    p.a = *reinterpret_cast<const uint4*>(row + k0 + 16 * hi);
    p.b = *reinterpret_cast<const uint4*>(row + k0 + 16 * hi + 8);
    return __builtin_bit_cast(v16bf, p);
}

// B fragment: 32x16 bf16 where B(k,n) = W[n_base+n][k]; per-lane column n = lane&15
// elements 0..15 -> K = k0 + 16*hi + e  (16 contiguous halves = 32 bytes)
static __device__ __forceinline__ v16bf load_b(const unsigned short* __restrict__ wrow,
                                               int k0, int hi) {
    Pack32 p;
    const uint4* q = reinterpret_cast<const uint4*>(wrow + k0 + 16 * hi);
    p.a = q[0];
    p.b = q[1];
    return __builtin_bit_cast(v16bf, p);
}

static __device__ __forceinline__ v8f wmma_bf16(v16bf a, v16bf b, v8f c) {
    return __builtin_amdgcn_wmma_f32_16x16x32_bf16(
        /*neg_a=*/false, a, /*neg_b=*/false, b,
        /*c_mod=*/(short)0, c, /*reuse_a=*/false, /*reuse_b=*/false);
}

// ---------------- embedding gather + bf16 cast: x[S][B][E] ----------------
__global__ __launch_bounds__(256) void embed_cast_kernel(
    const int* __restrict__ seq, const float* __restrict__ emb,
    unsigned short* __restrict__ xbf) {
    int gid = blockIdx.x * 256 + threadIdx.x;   // S_*B_*E_/8 threads
    int e8  = gid & (E_ / 8 - 1);               // 0..63
    int tb  = gid >> 6;                         // t*B_ + b
    int t   = tb >> 6;                          // B_ = 64
    int b   = tb & (B_ - 1);
    int tok = seq[b * S_ + t];
    const float4* src = reinterpret_cast<const float4*>(emb + (size_t)tok * E_ + e8 * 8);
    float4 f0 = src[0], f1 = src[1];
    uint4 o;
    o.x = (unsigned)f2bf(f0.x) | ((unsigned)f2bf(f0.y) << 16);
    o.y = (unsigned)f2bf(f0.z) | ((unsigned)f2bf(f0.w) << 16);
    o.z = (unsigned)f2bf(f1.x) | ((unsigned)f2bf(f1.y) << 16);
    o.w = (unsigned)f2bf(f1.z) | ((unsigned)f2bf(f1.w) << 16);
    *reinterpret_cast<uint4*>(xbf + (size_t)tb * E_ + e8 * 8) = o;
}

// ---------------- f32 -> bf16 weight cast ----------------
__global__ __launch_bounds__(256) void cast_bf16_kernel(
    const float* __restrict__ src, unsigned short* __restrict__ dst, int n8) {
    int gid = blockIdx.x * 256 + threadIdx.x;
    if (gid >= n8) return;
    const float4* s = reinterpret_cast<const float4*>(src + (size_t)gid * 8);
    float4 f0 = s[0], f1 = s[1];
    uint4 o;
    o.x = (unsigned)f2bf(f0.x) | ((unsigned)f2bf(f0.y) << 16);
    o.y = (unsigned)f2bf(f0.z) | ((unsigned)f2bf(f0.w) << 16);
    o.z = (unsigned)f2bf(f1.x) | ((unsigned)f2bf(f1.y) << 16);
    o.w = (unsigned)f2bf(f1.z) | ((unsigned)f2bf(f1.w) << 16);
    *reinterpret_cast<uint4*>(dst + (size_t)gid * 8) = o;
}

// ---------------- h0 = 0 ----------------
__global__ __launch_bounds__(256) void init_h_kernel(float* __restrict__ hf,
                                                     unsigned short* __restrict__ hb) {
    int i = blockIdx.x * 256 + threadIdx.x;
    if (i < B_ * H_) { hf[i] = 0.0f; hb[i] = 0; }
}

// ---------------- one GRU timestep (bf16 WMMA, fused input+recurrent GEMM) ----
// grid = 16 blocks x 256 threads = 128 wave32; one wave per 32x16 output tile
// (two 16x16 M-tiles share every B (weight) fragment -> 10 loads per 6 WMMAs).
__global__ __launch_bounds__(256) void gru_step_kernel(
    const unsigned short* __restrict__ x,      // [S][B][E] bf16
    const unsigned short* __restrict__ wih,    // [3H][E]  bf16 (rows: r, z, n)
    const unsigned short* __restrict__ whh,    // [3H][H]  bf16
    const float* __restrict__ b_ih, const float* __restrict__ b_hh,
    const float* __restrict__ h_prev_f, const unsigned short* __restrict__ h_prev_b,
    float* __restrict__ h_next_f, unsigned short* __restrict__ h_next_b,
    int t) {
    const int lane = threadIdx.x & 31;
    const int wave = threadIdx.x >> 5;
    const int tile = blockIdx.x * 8 + wave;    // 0..127
    const int mt   = tile & 1;                 // 32-row half of the batch
    const int nt   = tile >> 1;                // column tile 0..63
    const int j0   = nt * 16;
    const int hi   = lane >> 4;
    const int ln   = lane & 15;
    const int m0   = 32 * mt;

    v8f acc_r0 = {}, acc_z0 = {}, acc_xn0 = {}, acc_hn0 = {};
    v8f acc_r1 = {}, acc_z1 = {}, acc_xn1 = {}, acc_hn1 = {};

    // ---- input contribution: gx = x_t @ w_ih^T   (K = E = 512) ----
    const unsigned short* arow0 = x + ((size_t)t * B_ + (m0 + ln)) * E_;
    const unsigned short* arow1 = arow0 + (size_t)16 * E_;
    const unsigned short* wr0   = wih + (size_t)(j0 + ln) * E_;
    const unsigned short* wz0   = wih + (size_t)(H_ + j0 + ln) * E_;
    const unsigned short* wn0   = wih + (size_t)(2 * H_ + j0 + ln) * E_;
    #pragma unroll 4
    for (int kb = 0; kb < E_ / 32; ++kb) {
        const int k0 = kb * 32;
        v16bf a0 = load_a(arow0, k0, hi);
        v16bf a1 = load_a(arow1, k0, hi);
        v16bf br = load_b(wr0, k0, hi);
        v16bf bz = load_b(wz0, k0, hi);
        v16bf bn = load_b(wn0, k0, hi);
        acc_r0  = wmma_bf16(a0, br, acc_r0);
        acc_r1  = wmma_bf16(a1, br, acc_r1);
        acc_z0  = wmma_bf16(a0, bz, acc_z0);
        acc_z1  = wmma_bf16(a1, bz, acc_z1);
        acc_xn0 = wmma_bf16(a0, bn, acc_xn0);
        acc_xn1 = wmma_bf16(a1, bn, acc_xn1);
    }

    // ---- recurrent contribution: gh = h_prev @ w_hh^T   (K = H = 1024) ----
    const unsigned short* hrow0 = h_prev_b + (size_t)(m0 + ln) * H_;
    const unsigned short* hrow1 = hrow0 + (size_t)16 * H_;
    const unsigned short* vr0   = whh + (size_t)(j0 + ln) * H_;
    const unsigned short* vz0   = whh + (size_t)(H_ + j0 + ln) * H_;
    const unsigned short* vn0   = whh + (size_t)(2 * H_ + j0 + ln) * H_;
    #pragma unroll 4
    for (int kb = 0; kb < H_ / 32; ++kb) {
        const int k0 = kb * 32;
        v16bf a0 = load_a(hrow0, k0, hi);
        v16bf a1 = load_a(hrow1, k0, hi);
        v16bf br = load_b(vr0, k0, hi);
        v16bf bz = load_b(vz0, k0, hi);
        v16bf bn = load_b(vn0, k0, hi);
        acc_r0  = wmma_bf16(a0, br, acc_r0);
        acc_r1  = wmma_bf16(a1, br, acc_r1);
        acc_z0  = wmma_bf16(a0, bz, acc_z0);
        acc_z1  = wmma_bf16(a1, bz, acc_z1);
        acc_hn0 = wmma_bf16(a0, bn, acc_hn0);
        acc_hn1 = wmma_bf16(a1, bn, acc_hn1);
    }

    // ---- gates (C/D layout: VGPR i -> M = m0 + 16*half + 8*hi + i, N = j0 + ln) ----
    const int   j   = j0 + ln;
    const float brr = b_ih[j]          + b_hh[j];
    const float bzz = b_ih[H_ + j]     + b_hh[H_ + j];
    const float bin = b_ih[2 * H_ + j];
    const float bhn = b_hh[2 * H_ + j];
    #pragma unroll
    for (int half = 0; half < 2; ++half) {
        const v8f& ar = half ? acc_r1 : acc_r0;
        const v8f& az = half ? acc_z1 : acc_z0;
        const v8f& ax = half ? acc_xn1 : acc_xn0;
        const v8f& ah = half ? acc_hn1 : acc_hn0;
        const int mbase = m0 + 16 * half + 8 * hi;
        #pragma unroll
        for (int i = 0; i < 8; ++i) {
            const int m = mbase + i;
            float r  = sigmoidf_(ar[i] + brr);
            float z  = sigmoidf_(az[i] + bzz);
            float n  = tanhf(ax[i] + bin + r * (ah[i] + bhn));
            float hp = h_prev_f[(size_t)m * H_ + j];
            float hn = (1.0f - z) * n + z * hp;
            h_next_f[(size_t)m * H_ + j] = hn;
            h_next_b[(size_t)m * H_ + j] = f2bf(hn);
        }
    }
}

// ---------------- final FC: logits[b,c] = h_last[b,:] . fc_w[c,:] + fc_b[c] ----------------
__global__ __launch_bounds__(128) void fc_kernel(
    const float* __restrict__ h, const float* __restrict__ fc_w,
    const float* __restrict__ fc_b, float* __restrict__ out) {
    int idx = threadIdx.x;          // 128 = 64 batches * 2 classes
    int b = idx >> 1, c = idx & 1;
    float acc = fc_b[c];
    const float* hr = h + (size_t)b * H_;
    const float* wr = fc_w + (size_t)c * H_;
    #pragma unroll 8
    for (int k = 0; k < H_; ++k) acc += hr[k] * wr[k];
    out[b * 2 + c] = acc;
}

extern "C" void kernel_launch(void* const* d_in, const int* in_sizes, int n_in,
                              void* d_out, int out_size, void* d_ws, size_t ws_size,
                              hipStream_t stream) {
    const int*   seq  = (const int*)d_in[0];
    const float* emb  = (const float*)d_in[1];
    const float* w_ih = (const float*)d_in[2];
    const float* w_hh = (const float*)d_in[3];
    const float* b_ih = (const float*)d_in[4];
    const float* b_hh = (const float*)d_in[5];
    const float* fc_w = (const float*)d_in[6];
    const float* fc_b = (const float*)d_in[7];
    float* out = (float*)d_out;

    // workspace carve-up (~44 MB total)
    char* ws = (char*)d_ws;
    unsigned short* x_bf   = (unsigned short*)ws;                 // S*B*E bf16 = 32 MB
    ws += (size_t)S_ * B_ * E_ * sizeof(unsigned short);
    unsigned short* wih_bf = (unsigned short*)ws;                 // 3H*E bf16 = 3 MB
    ws += (size_t)3 * H_ * E_ * sizeof(unsigned short);
    unsigned short* whh_bf = (unsigned short*)ws;                 // 3H*H bf16 = 6 MB
    ws += (size_t)3 * H_ * H_ * sizeof(unsigned short);
    float* h_f = (float*)ws;                                      // 2 x [B][H] f32
    ws += (size_t)2 * B_ * H_ * sizeof(float);
    unsigned short* h_b = (unsigned short*)ws;                    // 2 x [B][H] bf16
    ws += (size_t)2 * B_ * H_ * sizeof(unsigned short);

    // 1) embedding gather -> bf16 x[S][B][E]
    embed_cast_kernel<<<(S_ * B_ * E_ / 8) / 256, 256, 0, stream>>>(seq, emb, x_bf);
    // 2) weight casts
    cast_bf16_kernel<<<(3 * H_ * E_ / 8 + 255) / 256, 256, 0, stream>>>(w_ih, wih_bf, 3 * H_ * E_ / 8);
    cast_bf16_kernel<<<(3 * H_ * H_ / 8 + 255) / 256, 256, 0, stream>>>(w_hh, whh_bf, 3 * H_ * H_ / 8);
    // 3) h0 = 0 (buffer 0)
    init_h_kernel<<<(B_ * H_ + 255) / 256, 256, 0, stream>>>(h_f, h_b);

    // 4) 512 dependent GRU steps, ping-pong h buffers
    for (int t = 0; t < S_; ++t) {
        const int cur = t & 1, nxt = (t + 1) & 1;
        gru_step_kernel<<<16, 256, 0, stream>>>(
            x_bf, wih_bf, whh_bf, b_ih, b_hh,
            h_f + (size_t)cur * B_ * H_, h_b + (size_t)cur * B_ * H_,
            h_f + (size_t)nxt * B_ * H_, h_b + (size_t)nxt * B_ * H_, t);
    }

    // 5) final classifier; after 512 steps h_last is in buffer (512 & 1) == 0
    fc_kernel<<<1, 128, 0, stream>>>(h_f, fc_w, fc_b, out);
}
